// MyRNNModel_28484223107418
// MI455X (gfx1250) — compile-verified, matrix operand order checked by reading, not run on hardware
//
#include <hip/hip_runtime.h>
#include <hip/hip_bf16.h>

typedef __bf16 bf16_t;
typedef __attribute__((ext_vector_type(16))) __bf16 v16bf;
typedef __attribute__((ext_vector_type(8)))  float  v8f;
typedef __attribute__((ext_vector_type(4)))  float  v4f;

// Problem dims (from reference)
constexpr int B_ = 4096, W_ = 50, F_ = 16, C_ = 8, V_ = 100000, D_ = 32, H_ = 256;
constexpr int KX   = F_ + C_ * D_;   // 272 input features
constexpr int KPAD = 288;            // x part zero-padded to multiple of 32
constexpr int KTOT = KPAD + H_;      // 544 = 17 * 32  (WMMA K chunks)
constexpr int G4   = 4 * H_;         // 1024 gate outputs
constexpr int TILE_B  = 32;          // batch rows per block
constexpr int THREADS = 512;         // 16 waves: (2 m-tiles) x (8 j-chunks of 32)

__device__ __forceinline__ v8f splat8(float x) {
    v8f r;
#pragma unroll
    for (int i = 0; i < 8; ++i) r[i] = x;
    return r;
}

__device__ __forceinline__ float fast_rcp(float x) { return __builtin_amdgcn_rcpf(x); }

// Branchless sigmoid / tanh (no EXEC divergence on the per-step critical path)
__device__ __forceinline__ float sigm(float x) {
    return fast_rcp(1.0f + __expf(-x));
}
__device__ __forceinline__ float tanh_fast(float x) {
    float xc = fminf(fmaxf(x, -15.0f), 15.0f);
    float t  = __expf(2.0f * xc);
    return (t - 1.0f) * fast_rcp(t + 1.0f);
}

// Load 16 contiguous floats, convert to 16 bf16, store as one 32B LDS chunk
__device__ __forceinline__ void cvt16_to_lds(const float* __restrict__ src,
                                             bf16_t* __restrict__ dst) {
    v4f e0 = *(const v4f*)(src + 0);
    v4f e1 = *(const v4f*)(src + 4);
    v4f e2 = *(const v4f*)(src + 8);
    v4f e3 = *(const v4f*)(src + 12);
    v16bf pk;
#pragma unroll
    for (int i = 0; i < 4; ++i) {
        pk[i]      = (bf16_t)e0[i];
        pk[4 + i]  = (bf16_t)e1[i];
        pk[8 + i]  = (bf16_t)e2[i];
        pk[12 + i] = (bf16_t)e3[i];
    }
    *(v16bf*)dst = pk;
}

// Pack [w_ih | zero pad | w_hh] -> bf16 Wcomb (G4 x KTOT), and bias = b_ih + b_hh
__global__ void prep_weights(const float* __restrict__ w_ih,
                             const float* __restrict__ w_hh,
                             const float* __restrict__ b_ih,
                             const float* __restrict__ b_hh,
                             bf16_t* __restrict__ wcomb,
                             float*  __restrict__ bias) {
    int idx = blockIdx.x * blockDim.x + threadIdx.x;
    if (idx < G4 * KTOT) {
        int n = idx / KTOT, k = idx % KTOT;
        float v = 0.0f;
        if (k < KX)         v = w_ih[(size_t)n * KX + k];
        else if (k >= KPAD) v = w_hh[(size_t)n * H_ + (k - KPAD)];
        wcomb[idx] = (bf16_t)v;
    }
    if (idx < G4) bias[idx] = b_ih[idx] + b_hh[idx];
}

__global__ __launch_bounds__(THREADS, 1)
void lstm_fused(const float* __restrict__ numeric,   // (B, W*F)
                const int*   __restrict__ cats,      // (W, C, B)
                const float* __restrict__ emb,       // (C, V, D)
                const bf16_t* __restrict__ wcomb,    // (G4, KTOT) bf16
                const float* __restrict__ bias,      // (G4)
                const float* __restrict__ w_fc,      // (H)
                const float* __restrict__ b_fc,      // (1)
                float* __restrict__ out) {           // (B)
    __shared__ __attribute__((aligned(32))) bf16_t xh[TILE_B * KTOT];

    const int tid   = threadIdx.x;
    const int b0    = blockIdx.x * TILE_B;
    const int wave  = tid >> 5;
    const int lane  = tid & 31;
    const int lhalf = lane >> 4;      // K-half selector for A/B fragments
    const int l16   = lane & 15;
    const int m0 = (wave & 1) * 16;   // batch-row tile within block
    const int j0 = (wave >> 1) * 32;  // hidden-column chunk [j0, j0+32)

    // Warm L2 with the packed weights (global_prefetch_b8)
    for (size_t off = (size_t)tid * 64; off < (size_t)G4 * KTOT; off += (size_t)THREADS * 64)
        __builtin_prefetch(&wcomb[off], 0, 1);

    // Zero LDS (h part must start at 0; pad must stay 0)
    for (int i = tid; i < TILE_B * KTOT; i += THREADS) xh[i] = (bf16_t)0.0f;

    // Persistent per-thread state: bias regs and cell state (never touches LDS)
    float breg[4][2];
#pragma unroll
    for (int g = 0; g < 4; ++g)
#pragma unroll
        for (int nt = 0; nt < 2; ++nt)
            breg[g][nt] = bias[g * H_ + j0 + nt * 16 + l16];

    v8f cstate[2];
#pragma unroll
    for (int nt = 0; nt < 2; ++nt) cstate[nt] = splat8(0.0f);

    v8f acc[4][2];

    __syncthreads();

    for (int w = 0; w < W_; ++w) {
        // ---- Gather x_t into xh[:, 0..KX): 32 rows x 8 cats x 2 halves ----
        {
            int b    = tid >> 4;        // 0..31
            int cc   = (tid >> 1) & 7;  // 0..7
            int half = tid & 1;         // 0..1
            int idx = cats[(size_t)w * C_ * B_ + (size_t)cc * B_ + (b0 + b)];
            const float* ep = emb + ((size_t)cc * V_ + (size_t)idx) * D_ + half * 16;
            cvt16_to_lds(ep, &xh[b * KTOT + F_ + cc * D_ + half * 16]);

            if (tid < TILE_B) {
                const float* np = numeric + (size_t)(b0 + tid) * (W_ * F_) + (size_t)w * F_;
                cvt16_to_lds(np, &xh[tid * KTOT]);
            }
        }
        __syncthreads();

        // ---- gates = [x_t | h] @ Wcomb^T + bias  (bf16 WMMA, f32 accum) ----
#pragma unroll
        for (int g = 0; g < 4; ++g)
#pragma unroll
            for (int nt = 0; nt < 2; ++nt)
                acc[g][nt] = splat8(breg[g][nt]);

#pragma unroll 1
        for (int k0 = 0; k0 < KTOT; k0 += 32) {
            // A fragment: 16x32 bf16 tile of xh; lane l = row (l&15), K-half (l>=16)
            v16bf afrag = *(const v16bf*)&xh[(m0 + l16) * KTOT + k0 + lhalf * 16];
#pragma unroll
            for (int g = 0; g < 4; ++g) {
#pragma unroll
                for (int nt = 0; nt < 2; ++nt) {
                    int n = g * H_ + j0 + nt * 16 + l16;   // gate output column
                    // B fragment: 32x16, B = Wcomb^T; lane l = column n, K-half (l>=16)
                    v16bf bfrag = *(const v16bf*)&wcomb[(size_t)n * KTOT + k0 + lhalf * 16];
                    acc[g][nt] = __builtin_amdgcn_wmma_f32_16x16x32_bf16(
                        false, afrag, false, bfrag, (short)0, acc[g][nt], false, false);
                }
            }
        }
        __syncthreads();  // everyone done reading h before rewriting it

        // ---- Elementwise LSTM cell update; c stays in VGPRs, h -> LDS bf16 ----
#pragma unroll
        for (int nt = 0; nt < 2; ++nt) {
#pragma unroll
            for (int v = 0; v < 8; ++v) {
                float xi = acc[0][nt][v];
                float xf = acc[1][nt][v];
                float xg = acc[2][nt][v];
                float xo = acc[3][nt][v];
                float c  = sigm(xf) * cstate[nt][v] + sigm(xi) * tanh_fast(xg);
                cstate[nt][v] = c;
                float h = sigm(xo) * tanh_fast(c);
                int m = v + 8 * lhalf;  // C/D layout: vgpr v, lanes>=16 hold M=v+8
                xh[(m0 + m) * KTOT + KPAD + j0 + nt * 16 + l16] = (bf16_t)h;
            }
        }
        __syncthreads();
    }

    // ---- Final FC: sigmoid(relu(h_T) @ w_fc^T + b_fc) ----
    if (tid < TILE_B) {
        const bf16_t* hrow = &xh[tid * KTOT + KPAD];
        float s = 0.0f;
        for (int j = 0; j < H_; ++j) {
            float hv = (float)hrow[j];
            s += fmaxf(hv, 0.0f) * w_fc[j];
        }
        out[b0 + tid] = sigm(s + b_fc[0]);
    }
}

extern "C" void kernel_launch(void* const* d_in, const int* in_sizes, int n_in,
                              void* d_out, int out_size, void* d_ws, size_t ws_size,
                              hipStream_t stream) {
    const float* numeric = (const float*)d_in[0];
    const int*   cats    = (const int*)d_in[1];
    const float* emb     = (const float*)d_in[2];
    const float* w_ih    = (const float*)d_in[3];
    const float* w_hh    = (const float*)d_in[4];
    const float* b_ih    = (const float*)d_in[5];
    const float* b_hh    = (const float*)d_in[6];
    const float* w_fc    = (const float*)d_in[7];
    const float* b_fc    = (const float*)d_in[8];
    float* out = (float*)d_out;

    bf16_t* wcomb = (bf16_t*)d_ws;
    float*  bias  = (float*)((char*)d_ws + (size_t)G4 * KTOT * sizeof(bf16_t));

    int prepTotal = G4 * KTOT;
    prep_weights<<<(prepTotal + 255) / 256, 256, 0, stream>>>(w_ih, w_hh, b_ih, b_hh, wcomb, bias);
    lstm_fused<<<B_ / TILE_B, THREADS, 0, stream>>>(numeric, cats, emb, wcomb, bias, w_fc, b_fc, out);
}
